// GGMLLayer_47648367182529
// MI455X (gfx1250) — compile-verified, matrix operand order checked by reading, not run on hardware
//
#include <hip/hip_runtime.h>

// ---- vector types ----
typedef _Float16 h8   __attribute__((ext_vector_type(8)));
typedef _Float16 v16h __attribute__((ext_vector_type(16)));
typedef float    v8f  __attribute__((ext_vector_type(8)));
typedef float    f4   __attribute__((ext_vector_type(4)));
typedef int      i4   __attribute__((ext_vector_type(4)));

union ABFrag { v16h v; h8 h[2]; _Float16 e[16]; };

#define KSTEP   32            // one WMMA K == one Q8_0 scale block
#define TILE_M  128           // workgroup M tile (wave covers 64 rows)
#define TILE_N  128           // workgroup N tile (wave covers 32 cols)
#define LDS_STR 40            // halfs per row: 80 B, 16-B aligned

static __device__ __forceinline__ v8f wmma_f16(v16h a, v16h b, v8f c) {
    // (neg_a, A, neg_b, B, c_mod, C, reuse_a, reuse_b)
    return __builtin_amdgcn_wmma_f32_16x16x32_f16(false, a, false, b,
                                                  (short)0, c, false, false);
}

__global__ __launch_bounds__(256)
void q8_linear_wmma(const float* __restrict__ x,     // [M, K] fp32
                    const int*   __restrict__ qw,    // [O, K] int8 stored as int32
                    const float* __restrict__ scales,// [O, K/32]
                    const float* __restrict__ bias,  // [O]
                    float*       __restrict__ out,   // [M, O]
                    int M, int K, int O)
{
    __shared__ _Float16 xs[TILE_M * LDS_STR];        // 128 x 32 f16 (+pad) = 10 KB

    const int tid    = threadIdx.x;
    const int lane   = tid & 31;
    const int wave   = tid >> 5;
    const int laneLo = lane & 15;
    const int laneHi = lane >> 4;            // 0/1 -> K sub-chunk select
    const int waveM  = wave >> 2;            // 0..1 -> 64-row half of tile
    const int waveN  = wave & 3;             // 0..3 -> 32-col quarter of tile

    const int mBase = blockIdx.y * TILE_M;
    const int nBase = blockIdx.x * TILE_N;

    // --- x-tile loaders: 128 rows x 32 cols fp32; thread -> 16 f32 ---
    const int xRow = tid >> 1;               // 0..127
    const int xCol = (tid & 1) * 16;         // 0 or 16
    const float* xRowPtr = x + (size_t)(mBase + xRow) * (size_t)K + xCol;

    const int KB = K >> 5;                   // scale blocks per row

    // --- per-wave weight fragment rows (B operand: N = o, K contiguous) ---
    const int oa = nBase + waveN * 32 + laneLo;   // fn=0 column
    const int ob = oa + 16;                       // fn=1 column
    const int*   wRow0 = qw + (size_t)oa * (size_t)K + 8 * laneHi;
    const int*   wRow1 = qw + (size_t)ob * (size_t)K + 8 * laneHi;
    const float* sRow0 = scales + (size_t)oa * (size_t)KB;
    const float* sRow1 = scales + (size_t)ob * (size_t)KB;

    v8f acc[4][2];
    #pragma unroll
    for (int fm = 0; fm < 4; ++fm) {
        acc[fm][0] = (v8f){};
        acc[fm][1] = (v8f){};
    }

    // A fragment LDS base: row waveM*64 + fm*16 + laneLo, K chunk select 8*laneHi
    const _Float16* aBase = &xs[(waveM * 64 + laneLo) * LDS_STR] + 8 * laneHi;

    // ================= software-pipeline prologue: loads for kb = 0 ==========
    f4 xc0 = *(const f4*)(xRowPtr);
    f4 xc1 = *(const f4*)(xRowPtr + 4);
    f4 xc2 = *(const f4*)(xRowPtr + 8);
    f4 xc3 = *(const f4*)(xRowPtr + 12);
    i4 qc[2][4];
    qc[0][0] = *(const i4*)(wRow0);       qc[0][1] = *(const i4*)(wRow0 + 4);
    qc[0][2] = *(const i4*)(wRow0 + 16);  qc[0][3] = *(const i4*)(wRow0 + 20);
    qc[1][0] = *(const i4*)(wRow1);       qc[1][1] = *(const i4*)(wRow1 + 4);
    qc[1][2] = *(const i4*)(wRow1 + 16);  qc[1][3] = *(const i4*)(wRow1 + 20);
    float sc0 = sRow0[0];
    float sc1 = sRow1[0];

    for (int kb = 0; kb < K; kb += KSTEP) {
        // ---- stage x tile (regs loaded last iteration) fp32 -> f16 into LDS ----
        h8 h0, h1;
        #pragma unroll
        for (int j = 0; j < 4; ++j) {
            h0[j]     = (_Float16)xc0[j];
            h0[4 + j] = (_Float16)xc1[j];
            h1[j]     = (_Float16)xc2[j];
            h1[4 + j] = (_Float16)xc3[j];
        }
        __syncthreads();                                   // prev iter's LDS reads done
        *(h8*)&xs[xRow * LDS_STR + xCol]     = h0;         // ds_store_b128
        *(h8*)&xs[xRow * LDS_STR + xCol + 8] = h1;
        __syncthreads();                                   // tile visible

        // ---- issue NEXT iteration's global loads (in flight during WMMAs) ----
        const int kn = (kb + KSTEP < K) ? (kb + KSTEP) : 0;   // branchless wrap
        f4 xn0 = *(const f4*)(xRowPtr + kn);
        f4 xn1 = *(const f4*)(xRowPtr + kn + 4);
        f4 xn2 = *(const f4*)(xRowPtr + kn + 8);
        f4 xn3 = *(const f4*)(xRowPtr + kn + 12);
        i4 qn[2][4];
        qn[0][0] = *(const i4*)(wRow0 + kn);       qn[0][1] = *(const i4*)(wRow0 + kn + 4);
        qn[0][2] = *(const i4*)(wRow0 + kn + 16);  qn[0][3] = *(const i4*)(wRow0 + kn + 20);
        qn[1][0] = *(const i4*)(wRow1 + kn);       qn[1][1] = *(const i4*)(wRow1 + kn + 4);
        qn[1][2] = *(const i4*)(wRow1 + kn + 16);  qn[1][3] = *(const i4*)(wRow1 + kn + 20);
        const float sn0 = sRow0[kn >> 5];
        const float sn1 = sRow1[kn >> 5];
        __builtin_prefetch(wRow0 + kb + 2 * KSTEP, 0, 1);  // GL2 warm, 2 blocks ahead
        __builtin_prefetch(wRow1 + kb + 2 * KSTEP, 0, 1);

        // ---- A fragments from LDS (16-bit A layout) ----
        ABFrag a[4];
        #pragma unroll
        for (int fm = 0; fm < 4; ++fm) {
            const _Float16* ar = aBase + fm * 16 * LDS_STR;
            a[fm].h[0] = *(const h8*)(ar);                 // ds_load_b128
            a[fm].h[1] = *(const h8*)(ar + 16);
        }

        // ---- dequant current B (regs from last iteration) + 8 WMMAs ----
        #pragma unroll
        for (int fn = 0; fn < 2; ++fn) {
            const float s = fn ? sc1 : sc0;
            ABFrag b;
            #pragma unroll
            for (int j = 0; j < 4; ++j) {
                b.e[j]      = (_Float16)((float)qc[fn][0][j] * s);
                b.e[4 + j]  = (_Float16)((float)qc[fn][1][j] * s);
                b.e[8 + j]  = (_Float16)((float)qc[fn][2][j] * s);
                b.e[12 + j] = (_Float16)((float)qc[fn][3][j] * s);
            }
            #pragma unroll
            for (int fm = 0; fm < 4; ++fm)
                acc[fm][fn] = wmma_f16(a[fm].v, b.v, acc[fm][fn]);
        }

        // ---- rotate pipeline registers ----
        xc0 = xn0; xc1 = xn1; xc2 = xn2; xc3 = xn3;
        #pragma unroll
        for (int fn = 0; fn < 2; ++fn)
            #pragma unroll
            for (int c = 0; c < 4; ++c)
                qc[fn][c] = qn[fn][c];
        sc0 = sn0; sc1 = sn1;
    }

    // ---- epilogue: D layout: VGPR r -> M = r + 8*laneHi, N = laneLo ----
    const float biasA = bias[oa];
    const float biasB = bias[ob];

    #pragma unroll
    for (int fm = 0; fm < 4; ++fm) {
        const int mrow = mBase + waveM * 64 + fm * 16 + 8 * laneHi;
        #pragma unroll
        for (int r = 0; r < 8; ++r) {
            float* po = out + (size_t)(mrow + r) * (size_t)O + oa;
            po[0]  = acc[fm][0][r] + biasA;
            po[16] = acc[fm][1][r] + biasB;
        }
    }
}

extern "C" void kernel_launch(void* const* d_in, const int* in_sizes, int n_in,
                              void* d_out, int out_size, void* d_ws, size_t ws_size,
                              hipStream_t stream) {
    const float* x      = (const float*)d_in[0];
    const int*   qw     = (const int*)d_in[1];
    const float* scales = (const float*)d_in[2];
    const float* bias   = (const float*)d_in[3];
    float*       out    = (float*)d_out;

    const int O = in_sizes[3];                 // 16384
    const int K = in_sizes[1] / O;             // 4096
    const int M = in_sizes[0] / K;             // B*S = 4096

    dim3 grid(O / TILE_N, M / TILE_M);         // shapes divide tiles exactly
    q8_linear_wmma<<<grid, 256, 0, stream>>>(x, qw, scales, bias, out, M, K, O);
}
